// LinearAttention_68410239091204
// MI455X (gfx1250) — compile-verified, matrix operand order checked by reading, not run on hardware
//
#include <hip/hip_runtime.h>
#include <cstdint>

typedef __attribute__((ext_vector_type(16))) _Float16 v16h;
typedef __attribute__((ext_vector_type(8)))  _Float16 h8;
typedef __attribute__((ext_vector_type(8)))  float    v8f;

#define HEADS      8
#define HEAD_DIM   32
#define HID        256
#define DIM        256
#define NPIX       4096          // 64*64
#define BATCH      16
#define NROWS      (BATCH * NPIX) // 65536
#define QKV_N      768
#define RMS_SCALE  16.0f
#define ATTN_SCALE 0.1767766952966369f   // 1/sqrt(32)

// ---------------------------------------------------------------------------
// Kernel 1: fused  xn = x*(gamma_in+1)*16 ;  qkv = xn @ w_qkv + b_qkv
//           q -> per-pixel softmax over the 32 head-dims (in-register)
//           q_soft, k, v stored as f16 (L2-resident intermediates)
// Block: 256 thr (8 waves). Block tile 128x64, wave tile 32x32 (2x2 WMMA).
// LDS staging is packed so every thread commits 16-byte ds_store_b128s.
// ---------------------------------------------------------------------------
__launch_bounds__(256)
__global__ void qkv_kernel(const float* __restrict__ x,
                           const float* __restrict__ gamma_in,
                           const float* __restrict__ w_qkv,
                           const float* __restrict__ b_qkv,
                           _Float16* __restrict__ qs,
                           _Float16* __restrict__ ks,
                           _Float16* __restrict__ vs)
{
    __shared__ __align__(16) _Float16 Asl[128][40];  // [row][k] f16, padded
    __shared__ __align__(16) _Float16 Bst[64][40];   // [n][k]  f16 (transposed)
    __shared__ float scall[DIM];

    const int t     = threadIdx.x;
    const int lane  = t & 31;
    const int wid   = t >> 5;
    const int waveM = wid & 3;   // 0..3 -> 32-row group
    const int waveN = wid >> 2;  // 0..1 -> 32-col group
    const int ln    = lane & 15;
    const int lh    = lane >> 4;

    const int rowBase = blockIdx.x * 128;
    const int colBase = blockIdx.y * 64;

    scall[t] = (gamma_in[t] + 1.0f) * RMS_SCALE;  // fused input RMS scale
    __syncthreads();

    v8f zacc = {};
    v8f acc[2][2];
    acc[0][0] = zacc; acc[0][1] = zacc; acc[1][0] = zacc; acc[1][1] = zacc;

    const int lr  = t >> 1;             // A-load: row 0..127
    const int lp  = (t & 1) * 16;       // A-load: 16-float half of the K slice
    const int bnl = t & 63;             // B-load: n 0..63
    const int bkg = (t >> 6) * 8;       // B-load: 8-wide K group

    for (int k0 = 0; k0 < DIM; k0 += 32) {
        // ---- stage A tile (128x32): scale + f32->f16, two b128 LDS stores --
        {
            const float* src = x + (size_t)(rowBase + lr) * DIM + k0 + lp;
            if (k0 + 32 < DIM)
                __builtin_prefetch(src + 32, 0, 3);   // global_prefetch_b8
            #pragma unroll
            for (int half = 0; half < 2; ++half) {
                h8 pk;
                #pragma unroll
                for (int j = 0; j < 8; j += 4) {
                    float4 f = *(const float4*)(src + half * 8 + j);
                    const int c = k0 + lp + half * 8 + j;
                    pk[j + 0] = (_Float16)(f.x * scall[c + 0]);
                    pk[j + 1] = (_Float16)(f.y * scall[c + 1]);
                    pk[j + 2] = (_Float16)(f.z * scall[c + 2]);
                    pk[j + 3] = (_Float16)(f.w * scall[c + 3]);
                }
                *(h8*)&Asl[lr][lp + half * 8] = pk;
            }
        }
        // ---- stage B tile (32x64) transposed: one b128 LDS store/thread ----
        {
            h8 bv;
            #pragma unroll
            for (int j = 0; j < 8; ++j)
                bv[j] = (_Float16)w_qkv[(size_t)(k0 + bkg + j) * QKV_N + colBase + bnl];
            *(h8*)&Bst[bnl][bkg] = bv;
        }
        __syncthreads();

        // ---- fragments: two contiguous 8-half K-runs per lane (b128 loads) ----
        v16h bf[2];
        #pragma unroll
        for (int ct = 0; ct < 2; ++ct) {
            const int n = waveN * 32 + ct * 16 + ln;
            h8 blo = *(const h8*)&Bst[n][8 * lh];
            h8 bhi = *(const h8*)&Bst[n][16 + 8 * lh];
            bf[ct] = __builtin_shufflevector(blo, bhi,
                     0,1,2,3,4,5,6,7,8,9,10,11,12,13,14,15);
        }
        #pragma unroll
        for (int rt = 0; rt < 2; ++rt) {
            const int r = waveM * 32 + rt * 16 + ln;
            h8 alo = *(const h8*)&Asl[r][8 * lh];
            h8 ahi = *(const h8*)&Asl[r][16 + 8 * lh];
            v16h af = __builtin_shufflevector(alo, ahi,
                      0,1,2,3,4,5,6,7,8,9,10,11,12,13,14,15);
            #pragma unroll
            for (int ct = 0; ct < 2; ++ct) {
                acc[rt][ct] = __builtin_amdgcn_wmma_f32_16x16x32_f16(
                    false, af, false, bf[ct], (short)0, acc[rt][ct], false, false);
            }
        }
        __syncthreads();
    }

    // ---- epilogue: bias; q region gets feature softmax; store f16 ----
    const int colW = colBase + waveN * 32;           // multiple of 32
    const float bias0 = b_qkv[colW + ln];
    const float bias1 = b_qkv[colW + 16 + ln];
    const int region = colW >> 8;                    // 0=q 1=k 2=v

    if (region == 0) {
        #pragma unroll
        for (int rt = 0; rt < 2; ++rt) {
            #pragma unroll
            for (int vg = 0; vg < 8; ++vg) {
                float v0 = (acc[rt][0][vg] + bias0) * ATTN_SCALE;
                float v1 = (acc[rt][1][vg] + bias1) * ATTN_SCALE;
                float m = fmaxf(v0, v1);
                #pragma unroll
                for (int msk = 1; msk <= 8; msk <<= 1)
                    m = fmaxf(m, __shfl_xor(m, msk, 32));   // stays in 16-lane half
                float e0 = __expf(v0 - m);
                float e1 = __expf(v1 - m);
                float s = e0 + e1;
                #pragma unroll
                for (int msk = 1; msk <= 8; msk <<= 1)
                    s += __shfl_xor(s, msk, 32);
                const float inv = 1.0f / s;
                const int row = rowBase + waveM * 32 + rt * 16 + vg + 8 * lh;
                _Float16* dst = qs + (size_t)row * HID + colW;
                dst[ln]      = (_Float16)(e0 * inv);
                dst[16 + ln] = (_Float16)(e1 * inv);
            }
        }
    } else {
        _Float16* outp = (region == 1) ? ks : vs;
        const int cb = colW - region * 256;
        #pragma unroll
        for (int rt = 0; rt < 2; ++rt) {
            #pragma unroll
            for (int vg = 0; vg < 8; ++vg) {
                const int row = rowBase + waveM * 32 + rt * 16 + vg + 8 * lh;
                _Float16* dst = outp + (size_t)row * HID + cb;
                dst[ln]      = (_Float16)(acc[rt][0][vg] + bias0);
                dst[16 + ln] = (_Float16)(acc[rt][1][vg] + bias1);
            }
        }
    }
}

// ---------------------------------------------------------------------------
// Kernel 2: k-softmax stats per (batch, channel): max & sum-exp over the
// 4096-pixel sequence plus the 4 memory slots. Grid (16, 4), 256 thr.
// ---------------------------------------------------------------------------
__launch_bounds__(256)
__global__ void kstats_kernel(const _Float16* __restrict__ ks,
                              const float* __restrict__ mem_kv,
                              float* __restrict__ Mb,
                              float* __restrict__ Zb)
{
    __shared__ float red[4][64];
    __shared__ float mrow[64];
    const int t  = threadIdx.x;
    const int b  = blockIdx.x;
    const int cl = t & 63;
    const int c  = blockIdx.y * 64 + cl;
    const int rg = t >> 6;
    const size_t base = (size_t)b * NPIX * 256 + c;

    float m = -1e30f;
    for (int r = rg; r < NPIX; r += 4)
        m = fmaxf(m, (float)ks[base + (size_t)r * 256]);
    red[rg][cl] = m;
    __syncthreads();
    if (rg == 0) {
        m = fmaxf(fmaxf(red[0][cl], red[1][cl]), fmaxf(red[2][cl], red[3][cl]));
        const int h = c >> 5, d = c & 31;
        #pragma unroll
        for (int mm = 0; mm < 4; ++mm)
            m = fmaxf(m, mem_kv[(h * 4 + mm) * 32 + d]);
        mrow[cl] = m;
    }
    __syncthreads();
    m = mrow[cl];
    float s = 0.0f;
    for (int r = rg; r < NPIX; r += 4)
        s += __expf((float)ks[base + (size_t)r * 256] - m);
    __syncthreads();
    red[rg][cl] = s;
    __syncthreads();
    if (rg == 0) {
        s = red[0][cl] + red[1][cl] + red[2][cl] + red[3][cl];
        const int h = c >> 5, d = c & 31;
        #pragma unroll
        for (int mm = 0; mm < 4; ++mm)
            s += __expf(mem_kv[(h * 4 + mm) * 32 + d] - m);
        Mb[b * 256 + c] = m;
        Zb[b * 256 + c] = s;
    }
}

// ---------------------------------------------------------------------------
// Kernel 3: context partials ctx[d][e] += exp(k-M)*v over a 256-pixel chunk.
// Grid (128, 16), 256 thr; atomicAdd f32 partials.
// ---------------------------------------------------------------------------
__launch_bounds__(256)
__global__ void ctx_kernel(const _Float16* __restrict__ ks,
                           const _Float16* __restrict__ vs,
                           const float* __restrict__ Mb,
                           float* __restrict__ ctxp)
{
    __shared__ float ek[32][33];
    __shared__ float vv[32][33];
    __shared__ float ml[32];
    const int t  = threadIdx.x;
    const int bh = blockIdx.x;
    const int b  = bh >> 3, h = bh & 7;
    const int rowBase = b * NPIX + blockIdx.y * 256;

    if (t < 32) ml[t] = Mb[b * 256 + h * 32 + t];

    const int d  = t >> 3;
    const int e0 = (t & 7) * 4;
    float a0 = 0.f, a1 = 0.f, a2 = 0.f, a3 = 0.f;

    const int pixLd = t >> 3;
    const int cgLd  = (t & 7) * 4;

    for (int s0 = 0; s0 < 256; s0 += 32) {
        __syncthreads();
        const size_t roff = (size_t)(rowBase + s0 + pixLd) * 256 + h * 32 + cgLd;
        #pragma unroll
        for (int j = 0; j < 4; ++j) {
            const float kf = (float)ks[roff + j];
            ek[pixLd][cgLd + j] = __expf(kf - ml[cgLd + j]);
            vv[pixLd][cgLd + j] = (float)vs[roff + j];
        }
        __syncthreads();
        #pragma unroll 8
        for (int p = 0; p < 32; ++p) {
            const float ekd = ek[p][d];
            a0 += ekd * vv[p][e0 + 0];
            a1 += ekd * vv[p][e0 + 1];
            a2 += ekd * vv[p][e0 + 2];
            a3 += ekd * vv[p][e0 + 3];
        }
    }
    float* dst = ctxp + (size_t)bh * 1024 + d * 32 + e0;
    atomicAdd(dst + 0, a0);
    atomicAdd(dst + 1, a1);
    atomicAdd(dst + 2, a2);
    atomicAdd(dst + 3, a3);
}

// ---------------------------------------------------------------------------
// Kernel 3b: add memory-KV term, divide by Z, emit f16 context. 128 blocks.
// ---------------------------------------------------------------------------
__launch_bounds__(256)
__global__ void ctxfin_kernel(const float* __restrict__ ctxp,
                              const float* __restrict__ mem_kv,
                              const float* __restrict__ Mb,
                              const float* __restrict__ Zb,
                              _Float16* __restrict__ ctxh)
{
    const int bh = blockIdx.x;
    const int b  = bh >> 3, h = bh & 7;
    for (int i = threadIdx.x; i < 1024; i += 256) {
        const int d = i >> 5, e = i & 31;
        const float m = Mb[b * 256 + h * 32 + d];
        const float z = Zb[b * 256 + h * 32 + d];
        float acc = ctxp[(size_t)bh * 1024 + i];
        #pragma unroll
        for (int mm = 0; mm < 4; ++mm) {
            const float mk = mem_kv[(h * 4 + mm) * 32 + d];
            const float mv = mem_kv[1024 + (h * 4 + mm) * 32 + e];
            acc += __expf(mk - m) * mv;
        }
        ctxh[(size_t)bh * 1024 + i] = (_Float16)(acc / z);
    }
}

// ---------------------------------------------------------------------------
// Kernel 4a: W2[b][c][j] = sum_e ctx[b,h(c)][d(c)][e] * w_out[h*32+e][j]
// (folds block-diagonal context GEMM into the output projection weights)
// ---------------------------------------------------------------------------
__launch_bounds__(256)
__global__ void w2_kernel(const _Float16* __restrict__ ctxh,
                          const float* __restrict__ w_out,
                          _Float16* __restrict__ W2)
{
    __shared__ float crow[32];
    const int b = blockIdx.x, c = blockIdx.y;
    const int h = c >> 5, d = c & 31;
    const int t = threadIdx.x;
    if (t < 32) crow[t] = (float)ctxh[(size_t)(b * 8 + h) * 1024 + d * 32 + t];
    __syncthreads();
    float acc = 0.0f;
    #pragma unroll 8
    for (int e = 0; e < 32; ++e)
        acc += crow[e] * w_out[(size_t)(h * 32 + e) * 256 + t];
    W2[((size_t)b * 256 + c) * 256 + t] = (_Float16)acc;
}

// ---------------------------------------------------------------------------
// Kernel 5: y = q_soft @ W2[b] + b_out, * (gamma_out+1)*16.  WMMA GEMM.
// A-tile (pure f16 copy) is staged with GLOBAL_LOAD_ASYNC_TO_LDS_B128
// (ASYNCcnt-tracked, no VGPR round trip); drained with s_wait_asynccnt
// before the workgroup barrier.
// ---------------------------------------------------------------------------
__launch_bounds__(256)
__global__ void out_kernel(const _Float16* __restrict__ qs,
                           const _Float16* __restrict__ W2,
                           const float* __restrict__ b_out,
                           const float* __restrict__ gamma_out,
                           float* __restrict__ y)
{
    __shared__ __align__(16) _Float16 Asl[128][40];
    __shared__ __align__(16) _Float16 Bst[64][40];

    const int t     = threadIdx.x;
    const int lane  = t & 31;
    const int wid   = t >> 5;
    const int waveM = wid & 3;
    const int waveN = wid >> 2;
    const int ln    = lane & 15;
    const int lh    = lane >> 4;

    const int rowBase = blockIdx.x * 128;
    const int colBase = blockIdx.y * 64;
    const int b       = rowBase >> 12;            // / 4096
    const _Float16* Wb = W2 + (size_t)b * 256 * 256;

    v8f zacc = {};
    v8f acc[2][2];
    acc[0][0] = zacc; acc[0][1] = zacc; acc[1][0] = zacc; acc[1][1] = zacc;

    const int lr  = t >> 1;
    const int lp  = (t & 1) * 16;
    const int bnl = t & 63;
    const int bkg = (t >> 6) * 8;

    for (int k0 = 0; k0 < 256; k0 += 32) {
        // ---- A tile: async global->LDS copy (b128, ASYNCcnt tracked) ----
        {
            const _Float16* gsrc = qs + (size_t)(rowBase + lr) * 256 + k0 + lp;
            const uint32_t ldsA = (uint32_t)(uintptr_t)&Asl[lr][lp];
            const uint64_t ga   = (uint64_t)(uintptr_t)gsrc;
            asm volatile("global_load_async_to_lds_b128 %0, %1, off"
                         :: "v"(ldsA), "v"(ga) : "memory");
            asm volatile("global_load_async_to_lds_b128 %0, %1, off offset:16"
                         :: "v"(ldsA), "v"(ga) : "memory");
        }
        // ---- B tile (32x64) transposed: one b128 LDS store per thread ----
        {
            h8 bv;
            #pragma unroll
            for (int j = 0; j < 8; ++j)
                bv[j] = Wb[(size_t)(k0 + bkg + j) * 256 + colBase + bnl];
            *(h8*)&Bst[bnl][bkg] = bv;
        }
        asm volatile("s_wait_asynccnt 0x0" ::: "memory");  // drain this wave's async loads
        __syncthreads();

        v16h bf[2];
        #pragma unroll
        for (int ct = 0; ct < 2; ++ct) {
            const int n = waveN * 32 + ct * 16 + ln;
            h8 blo = *(const h8*)&Bst[n][8 * lh];
            h8 bhi = *(const h8*)&Bst[n][16 + 8 * lh];
            bf[ct] = __builtin_shufflevector(blo, bhi,
                     0,1,2,3,4,5,6,7,8,9,10,11,12,13,14,15);
        }
        #pragma unroll
        for (int rt = 0; rt < 2; ++rt) {
            const int r = waveM * 32 + rt * 16 + ln;
            h8 alo = *(const h8*)&Asl[r][8 * lh];
            h8 ahi = *(const h8*)&Asl[r][16 + 8 * lh];
            v16h af = __builtin_shufflevector(alo, ahi,
                      0,1,2,3,4,5,6,7,8,9,10,11,12,13,14,15);
            #pragma unroll
            for (int ct = 0; ct < 2; ++ct) {
                acc[rt][ct] = __builtin_amdgcn_wmma_f32_16x16x32_f16(
                    false, af, false, bf[ct], (short)0, acc[rt][ct], false, false);
            }
        }
        __syncthreads();
    }

    const int colW = colBase + waveN * 32;
    const float b0 = b_out[colW + ln];
    const float b1 = b_out[colW + 16 + ln];
    const float g0 = (gamma_out[colW + ln] + 1.0f) * RMS_SCALE;
    const float g1 = (gamma_out[colW + 16 + ln] + 1.0f) * RMS_SCALE;

    #pragma unroll
    for (int rt = 0; rt < 2; ++rt) {
        #pragma unroll
        for (int vg = 0; vg < 8; ++vg) {
            const int row = rowBase + waveM * 32 + rt * 16 + vg + 8 * lh;
            float* dst = y + (size_t)row * 256 + colW;
            dst[ln]      = (acc[rt][0][vg] + b0) * g0;
            dst[16 + ln] = (acc[rt][1][vg] + b1) * g1;
        }
    }
}

// ---------------------------------------------------------------------------
extern "C" void kernel_launch(void* const* d_in, const int* in_sizes, int n_in,
                              void* d_out, int out_size, void* d_ws, size_t ws_size,
                              hipStream_t stream)
{
    (void)in_sizes; (void)n_in; (void)out_size; (void)ws_size;

    const float* x         = (const float*)d_in[0];
    const float* gamma_in  = (const float*)d_in[1];
    const float* w_qkv     = (const float*)d_in[2];
    const float* b_qkv     = (const float*)d_in[3];
    const float* mem_kv    = (const float*)d_in[4];
    const float* w_out     = (const float*)d_in[5];
    const float* b_out     = (const float*)d_in[6];
    const float* gamma_out = (const float*)d_in[7];
    float* y = (float*)d_out;

    char* ws = (char*)d_ws;
    size_t off = 0;
    auto carve = [&](size_t bytes) -> char* {
        char* p = ws + off;
        off += (bytes + 255) & ~(size_t)255;
        return p;
    };
    _Float16* qs   = (_Float16*)carve((size_t)NROWS * HID * 2);   // 32 MB
    _Float16* ks   = (_Float16*)carve((size_t)NROWS * HID * 2);   // 32 MB
    _Float16* vs   = (_Float16*)carve((size_t)NROWS * HID * 2);   // 32 MB
    float*    Mb   = (float*)carve((size_t)BATCH * 256 * 4);
    float*    Zb   = (float*)carve((size_t)BATCH * 256 * 4);
    float*    ctxp = (float*)carve((size_t)BATCH * HEADS * 1024 * 4);
    _Float16* ctxh = (_Float16*)carve((size_t)BATCH * HEADS * 1024 * 2);
    _Float16* W2   = (_Float16*)carve((size_t)BATCH * 256 * 256 * 2);

    hipMemsetAsync(ctxp, 0, (size_t)BATCH * HEADS * 1024 * 4, stream);

    qkv_kernel  <<<dim3(NROWS / 128, QKV_N / 64), 256, 0, stream>>>(
        x, gamma_in, w_qkv, b_qkv, qs, ks, vs);
    kstats_kernel<<<dim3(BATCH, 4), 256, 0, stream>>>(ks, mem_kv, Mb, Zb);
    ctx_kernel  <<<dim3(BATCH * HEADS, 16), 256, 0, stream>>>(ks, vs, Mb, ctxp);
    ctxfin_kernel<<<BATCH * HEADS, 256, 0, stream>>>(ctxp, mem_kv, Mb, Zb, ctxh);
    w2_kernel   <<<dim3(BATCH, 256), 256, 0, stream>>>(ctxh, w_out, W2);
    out_kernel  <<<dim3(NROWS / 128, HID / 64), 256, 0, stream>>>(
        qs, W2, b_out, gamma_out, y);
}
